// OptimalTransport_41592463294658
// MI455X (gfx1250) — matrix operand order repaired and checked by reading, not compiled, not run on hardware
//
#include <hip/hip_runtime.h>
#include <hip/hip_bf16.h>
#include <cstdint>

// Problem constants (match reference)
#define BATCH   128
#define NFEAT   784            // W*H = 28*28
#define DDIM    512
#define SSUP    128
#define N_ITERS 10
#define EPS_F   2.2204460492503131e-16f  // float32(7/3 - 4/3 - 1) computed in f64

// GEMM tiling
#define NT      7              // 7 x 16 = 112 columns per block
#define NCOLS   112
#define NGROUPS 7              // 7 * 112 = 784
#define LDS_STRIDE 520         // halfwords per LDS row (512 + 8 pad)

typedef __attribute__((ext_vector_type(16))) __bf16          v16bf;
typedef __attribute__((ext_vector_type(8)))  float           v8f;
typedef __attribute__((ext_vector_type(16))) unsigned short  v16u;
typedef __attribute__((ext_vector_type(8)))  unsigned short  v8u;
typedef __attribute__((ext_vector_type(4)))  float           v4f;

static __device__ __forceinline__ unsigned short f2bf(float x) {
    // round-to-nearest-even fp32 -> bf16
    unsigned u = __float_as_uint(x);
    u += 0x7FFFu + ((u >> 16) & 1u);
    return (unsigned short)(u >> 16);
}

// Build a 16-halfword WMMA fragment from two 8-halfword chunks at p and
// p + second_off (A and B both pass second_off = 8: contiguous).
static __device__ __forceinline__ v16bf ldfrag(const unsigned short* p, int second_off) {
    v8u x0 = *(const v8u*)(p);
    v8u x1 = *(const v8u*)(p + second_off);
    v16u u = __builtin_shufflevector(x0, x1,
                 0, 1, 2, 3, 4, 5, 6, 7, 8, 9, 10, 11, 12, 13, 14, 15);
    return __builtin_bit_cast(v16bf, u);
}

// ---------------------------------------------------------------------------
// Kernel 1: L2-normalize target_support rows, emit bf16 codebook (S x D)
// grid = 128 blocks, block = 32 (one wave per row)
// ---------------------------------------------------------------------------
__global__ void ot_norm_target(const float* __restrict__ t,
                               unsigned short* __restrict__ tn) {
    const int s = blockIdx.x;          // row 0..127
    const int l = threadIdx.x;         // lane 0..31
    const float* row = t + (size_t)s * DDIM + l * 16;
    float v[16];
    float ss = 0.f;
#pragma unroll
    for (int i = 0; i < 4; ++i) {
        v4f x = *(const v4f*)(row + 4 * i);
        v[4 * i + 0] = x.x; v[4 * i + 1] = x.y;
        v[4 * i + 2] = x.z; v[4 * i + 3] = x.w;
        ss += x.x * x.x + x.y * x.y + x.z * x.z + x.w * x.w;
    }
#pragma unroll
    for (int m = 16; m >= 1; m >>= 1) ss += __shfl_xor(ss, m, 32);
    const float sc = rsqrtf(ss);
    unsigned short* dst = tn + (size_t)s * DDIM + l * 16;
    v8u o0, o1;
#pragma unroll
    for (int i = 0; i < 8; ++i) { o0[i] = f2bf(v[i] * sc); o1[i] = f2bf(v[8 + i] * sc); }
    *(v8u*)(dst)     = o0;
    *(v8u*)(dst + 8) = o1;
}

// ---------------------------------------------------------------------------
// Kernel 2: fused feature-normalize + bf16 WMMA GEMM + exp
//   K[b,s,n] = exp(20*<t_s, f_n> - 20)   written into d_out (scratch for now)
// grid = (BATCH, NGROUPS), block = 256 (8 waves), dynamic LDS = 112*520*2 B
// Each wave: one 16-row s-tile x 112 columns. Per k-step: one batched load
// clause (2 global + 14 ds) then 7 back-to-back WMMAs. Fully unrolled so the
// scheduler picks its own cross-iteration load lookahead within the register
// budget (explicit double-buffering overflowed the VGPR cap and spilled).
// ---------------------------------------------------------------------------
__global__ void __launch_bounds__(256)
ot_gemm_exp(const float* __restrict__ feats,
            const unsigned short* __restrict__ tn,
            float* __restrict__ Kout) {
    extern __shared__ unsigned short fs[];   // [NCOLS][LDS_STRIDE] bf16
    const int b    = blockIdx.x;
    const int n0   = blockIdx.y * NCOLS;
    const int tid  = threadIdx.x;
    const int w    = tid >> 5;               // wave 0..7
    const int l    = tid & 31;               // lane 0..31
    const int ln   = l & 15;
    const int half = l >> 4;

    // ---- Phase 1: stage 112 normalized feature rows into LDS as bf16 ----
    for (int r = w * 14; r < w * 14 + 14; ++r) {
        const float* row = feats + ((size_t)b * NFEAT + n0 + r) * DDIM + l * 16;
        float v[16];
        float ss = 0.f;
#pragma unroll
        for (int i = 0; i < 4; ++i) {
            v4f x = *(const v4f*)(row + 4 * i);
            v[4 * i + 0] = x.x; v[4 * i + 1] = x.y;
            v[4 * i + 2] = x.z; v[4 * i + 3] = x.w;
            ss += x.x * x.x + x.y * x.y + x.z * x.z + x.w * x.w;
        }
#pragma unroll
        for (int m = 16; m >= 1; m >>= 1) ss += __shfl_xor(ss, m, 32);
        const float sc = rsqrtf(ss);
        unsigned short* dst = fs + r * LDS_STRIDE + l * 16;
        v8u o0, o1;
#pragma unroll
        for (int i = 0; i < 8; ++i) { o0[i] = f2bf(v[i] * sc); o1[i] = f2bf(v[8 + i] * sc); }
        *(v8u*)(dst)     = o0;
        *(v8u*)(dst + 8) = o1;
    }
    __syncthreads();

    // ---- Phase 2: WMMA GEMM; wave w handles s-tile w ----
    const int srow = w * 16 + ln;            // A row this lane supplies
    // A fragment base (16x32 bf16): h -> K = k + half*16 + h (contiguous)
    const unsigned short* arow = tn + (size_t)srow * DDIM + half * 16;
    // B fragment base (32x16 bf16): lane column = t*16 + ln,
    //                               h -> K = k + half*16 + h (contiguous)
    const unsigned short* brow = fs + (size_t)ln * LDS_STRIDE + half * 16;

    v8f acc[NT] = {};

#pragma unroll
    for (int k = 0; k < DDIM; k += 32) {
        // Batched loads for this step: 2 global_load_b128 + 14 ds_load_b128.
        v16bf af = ldfrag(arow + k, 8);
        v16bf bf[NT];
#pragma unroll
        for (int t = 0; t < NT; ++t)
            bf[t] = ldfrag(brow + t * (16 * LDS_STRIDE) + k, 8);

        // 7 independent back-to-back WMMAs (distinct accumulators).
#pragma unroll
        for (int t = 0; t < NT; ++t)
            acc[t] = __builtin_amdgcn_wmma_f32_16x16x32_bf16(
                         false, af, false, bf[t], (short)0, acc[t], false, false);
    }

    // ---- Epilogue: K = exp(-GAMMA*(2 - 2*dot)) = exp(20*dot - 20) ----
#pragma unroll
    for (int t = 0; t < NT; ++t) {
        const int ncol = n0 + t * 16 + ln;
#pragma unroll
        for (int r = 0; r < 8; ++r) {
            const int s = w * 16 + r + 8 * half;    // C/D layout: VGPR r, half
            const float kv = __expf(20.0f * acc[t][r] - 20.0f);
            Kout[((size_t)b * SSUP + s) * NFEAT + ncol] = kv;
        }
    }
}

// ---------------------------------------------------------------------------
// Kernel 3: Sinkhorn iterations + Pi, in place on K (one block per batch).
// K stays L2-hot (401 KB/batch, 51 MB total vs 192 MB L2).
// ---------------------------------------------------------------------------
__global__ void ot_sinkhorn(float* __restrict__ Kmat) {
    __shared__ float u[SSUP];
    __shared__ float v[NFEAT];
    const int b   = blockIdx.x;
    const int tid = threadIdx.x;
    const int w   = tid >> 5;
    const int l   = tid & 31;
    float* Kb = Kmat + (size_t)b * SSUP * NFEAT;

    for (int s = tid; s < SSUP; s += 256) u[s] = 1.0f;
    __syncthreads();

    for (int it = 0; it <= N_ITERS; ++it) {
        // v = q / (u*K + eps)   (columns across threads -> coalesced per s)
        for (int n = tid; n < NFEAT; n += 256) {
            float r = 0.f;
            for (int s = 0; s < SSUP; ++s) r += u[s] * Kb[s * NFEAT + n];
            v[n] = (1.0f / NFEAT) / (r + EPS_F);
        }
        __syncthreads();
        if (it == N_ITERS) break;   // reference: 11 v-updates, 10 u-updates
        // u = p / (v*K^T + eps)  (one wave per row, wave-shuffle reduce)
        for (int s = w; s < SSUP; s += 8) {
            const float* Krow = Kb + s * NFEAT;
            float acc = 0.f;
            for (int n = l; n < NFEAT; n += 32) acc += v[n] * Krow[n];
#pragma unroll
            for (int m = 16; m >= 1; m >>= 1) acc += __shfl_xor(acc, m, 32);
            if (l == 0) u[s] = (1.0f / SSUP) / (acc + EPS_F);
        }
        __syncthreads();
    }

    // Pi = u^T * K * v, in place
    for (int s = 0; s < SSUP; ++s) {
        const float us = u[s];
        for (int n = tid; n < NFEAT; n += 256) {
            Kb[s * NFEAT + n] = us * Kb[s * NFEAT + n] * v[n];
        }
    }
}

// ---------------------------------------------------------------------------
extern "C" void kernel_launch(void* const* d_in, const int* in_sizes, int n_in,
                              void* d_out, int out_size, void* d_ws, size_t ws_size,
                              hipStream_t stream) {
    const float* feats = (const float*)d_in[0];   // (128, 28, 28, 512) f32
    const float* tsup  = (const float*)d_in[1];   // (128, 512) f32
    float* out = (float*)d_out;                   // (128, 128, 784) f32
    unsigned short* tn = (unsigned short*)d_ws;   // bf16 codebook, 128 KB

    ot_norm_target<<<SSUP, 32, 0, stream>>>(tsup, tn);

    dim3 grid(BATCH, NGROUPS);
    size_t shmem = (size_t)NCOLS * LDS_STRIDE * sizeof(unsigned short); // 116480 B
    ot_gemm_exp<<<grid, 256, shmem, stream>>>(feats, tn, out);

    ot_sinkhorn<<<BATCH, 256, 0, stream>>>(out);
}
